// AttentionBlock_16475494548203
// MI455X (gfx1250) — compile-verified, matrix operand order checked by reading, not run on hardware
//
#include <hip/hip_runtime.h>
#include <hip/hip_bf16.h>

// ---------------------------------------------------------------------------
// Transformer block for MI455X (gfx1250): bf16 WMMA GEMMs with f32 accumulate,
// flash-style attention with fused alibi, fp32 softmax/LN/residual epilogues.
// Double-buffered async global->LDS staging (ASYNCcnt), permlane reductions.
// ---------------------------------------------------------------------------

typedef __attribute__((ext_vector_type(16))) __bf16 v16bf;
typedef __attribute__((ext_vector_type(8)))  float  v8f;
typedef __attribute__((ext_vector_type(4)))  float  f4raw;   // raw 16B load type
typedef __attribute__((ext_vector_type(4)))  int    i4raw;   // for async builtin

union Frag {
  v16bf v;
  f4raw f[2];
};

__device__ __forceinline__ unsigned short f2bf(float x) {
  unsigned int u = __float_as_uint(x);
  u += 0x7fffu + ((u >> 16) & 1u);     // round-to-nearest-even
  return (unsigned short)(u >> 16);
}

// ---- async global -> LDS staging (gfx1250 GLOBAL_LOAD_ASYNC_TO_LDS_B128) ---
__device__ __forceinline__ void async_copy16(void* lds, const void* g) {
#if __has_builtin(__builtin_amdgcn_global_load_async_to_lds_b128)
  __builtin_amdgcn_global_load_async_to_lds_b128(
      (__attribute__((address_space(1))) i4raw*)g,
      (__attribute__((address_space(3))) i4raw*)lds, 0, 0);
#else
  *reinterpret_cast<f4raw*>(lds) = *reinterpret_cast<const f4raw*>(g);
#endif
}

__device__ __forceinline__ void async_wait() {
#if __has_builtin(__builtin_amdgcn_global_load_async_to_lds_b128)
#if __has_builtin(__builtin_amdgcn_s_wait_asynccnt)
  __builtin_amdgcn_s_wait_asynccnt(0);
#else
  asm volatile("s_wait_asynccnt 0" ::: "memory");
#endif
#endif
}

// ---- butterfly reductions within each 16-lane half (VALU permlane16) ------
__device__ __forceinline__ float perm16(float v, unsigned lo, unsigned hi) {
#if __has_builtin(__builtin_amdgcn_permlane16)
  return __uint_as_float(__builtin_amdgcn_permlane16(
      __float_as_uint(v), __float_as_uint(v), lo, hi, false, false));
#else
  (void)lo; (void)hi;
  return v;
#endif
}

__device__ __forceinline__ float redmax16(float v) {
#if __has_builtin(__builtin_amdgcn_permlane16)
  v = fmaxf(v, perm16(v, 0x67452301u, 0xEFCDAB89u));  // xor 1
  v = fmaxf(v, perm16(v, 0x54761032u, 0xDCFE98BAu));  // xor 2
  v = fmaxf(v, perm16(v, 0x32107654u, 0xBA98FEDCu));  // xor 4
  v = fmaxf(v, perm16(v, 0xFEDCBA98u, 0x76543210u));  // xor 8
#else
  v = fmaxf(v, __shfl_xor(v, 1, 32));
  v = fmaxf(v, __shfl_xor(v, 2, 32));
  v = fmaxf(v, __shfl_xor(v, 4, 32));
  v = fmaxf(v, __shfl_xor(v, 8, 32));
#endif
  return v;
}

__device__ __forceinline__ float redsum16(float v) {
#if __has_builtin(__builtin_amdgcn_permlane16)
  v += perm16(v, 0x67452301u, 0xEFCDAB89u);  // xor 1
  v += perm16(v, 0x54761032u, 0xDCFE98BAu);  // xor 2
  v += perm16(v, 0x32107654u, 0xBA98FEDCu);  // xor 4
  v += perm16(v, 0xFEDCBA98u, 0x76543210u);  // xor 8
#else
  v += __shfl_xor(v, 1, 32);
  v += __shfl_xor(v, 2, 32);
  v += __shfl_xor(v, 4, 32);
  v += __shfl_xor(v, 8, 32);
#endif
  return v;
}

// ------------------------------- converts ----------------------------------
__global__ __launch_bounds__(256) void f32_to_bf16_kernel(
    const float* __restrict__ in, unsigned short* __restrict__ out, int n) {
  int i = blockIdx.x * blockDim.x + threadIdx.x;
  int stride = gridDim.x * blockDim.x;
  for (; i < n; i += stride) out[i] = f2bf(in[i]);
}

// ------------------------------- GEMM --------------------------------------
// C[M,N] = epilogue( A[M,K] * W[N,K]^T )  (both operands K-major bf16)
// Block: 256 threads = 8 waves, 128x128 C tile, wave grid 2x4, 64x32 per wave.
// Double-buffered LDS tiles: async-stage K-block i+1 while computing block i.
#define BM 128
#define BN 128
#define BK 64
#define LDSS 72   // padded LDS row stride (halves) to avoid bank conflicts

// Epilogue modes (compile-time -> no dead pointers live in registers)
#define EPI_STORE_BF16      0   // outB = C
#define EPI_RESID_F32       1   // outF = resid + C
#define EPI_BIAS_GELU_BF16  2   // outB = gelu(C + bias)
#define EPI_BIAS_RESID_F32  3   // outF = resid + C + bias

template <int EPI>
__global__ __launch_bounds__(256) void gemm_bf16_kernel(
    const unsigned short* __restrict__ A,   // [M,K] bf16
    const unsigned short* __restrict__ W,   // [N,K] bf16
    int M, int N, int K,
    const float* __restrict__ bias,         // [N]
    const float* __restrict__ resid,        // [M,N] fp32
    float* __restrict__ outF,               // [M,N] fp32
    unsigned short* __restrict__ outB) {    // [M,N] bf16
  __shared__ __align__(16) unsigned short As[2][BM * LDSS];
  __shared__ __align__(16) unsigned short Bs[2][BN * LDSS];

  const int tid   = threadIdx.x;
  const int wave  = tid >> 5;
  const int lane  = tid & 31;
  const int lmod  = lane & 15;
  const int lhalf = lane >> 4;
  const int waveM = wave >> 2;              // 0..1
  const int waveN = wave & 3;               // 0..3
  const int tileM = blockIdx.y * BM;
  const int tileN = blockIdx.x * BN;

  auto stage = [&](int buf, int k0) {
#pragma unroll
    for (int c = 0; c < 4; ++c) {
      int idx = tid + c * 256;
      int r = idx >> 3;
      int kk = (idx & 7) << 3;
      async_copy16(&As[buf][r * LDSS + kk],
                   &A[(size_t)(tileM + r) * K + k0 + kk]);
      async_copy16(&Bs[buf][r * LDSS + kk],
                   &W[(size_t)(tileN + r) * K + k0 + kk]);
    }
  };

  const v8f vzero = {0.f, 0.f, 0.f, 0.f, 0.f, 0.f, 0.f, 0.f};
  v8f acc[4][2];
#pragma unroll
  for (int mt = 0; mt < 4; ++mt)
#pragma unroll
    for (int nt = 0; nt < 2; ++nt) acc[mt][nt] = vzero;

  stage(0, 0);
  async_wait();
  __syncthreads();

  int cur = 0;
  for (int k0 = 0; k0 < K; k0 += BK) {
    const bool more = (k0 + BK) < K;
    if (more) stage(cur ^ 1, k0 + BK);   // overlap next tile with compute

#pragma unroll
    for (int ks = 0; ks < BK; ks += 32) {
      // B fragments: lane holds column (N) of B, K across VGPRs
      Frag bfr[2];
#pragma unroll
      for (int nt = 0; nt < 2; ++nt) {
        const unsigned short* p =
            &Bs[cur][(waveN * 32 + nt * 16 + lmod) * LDSS + ks + lhalf * 8];
        bfr[nt].f[0] = *reinterpret_cast<const f4raw*>(p);
        bfr[nt].f[1] = *reinterpret_cast<const f4raw*>(p + 16);
      }
#pragma unroll
      for (int mt = 0; mt < 4; ++mt) {
        Frag af;  // lane holds row (M) of A, K across VGPRs
        const unsigned short* p =
            &As[cur][(waveM * 64 + mt * 16 + lmod) * LDSS + ks + lhalf * 8];
        af.f[0] = *reinterpret_cast<const f4raw*>(p);
        af.f[1] = *reinterpret_cast<const f4raw*>(p + 16);
        acc[mt][0] = __builtin_amdgcn_wmma_f32_16x16x32_bf16(
            false, af.v, false, bfr[0].v, (short)0, acc[mt][0], false, false);
        acc[mt][1] = __builtin_amdgcn_wmma_f32_16x16x32_bf16(
            false, af.v, false, bfr[1].v, (short)0, acc[mt][1], false, false);
      }
    }

    if (more) {
      async_wait();
      __syncthreads();
    }
    cur ^= 1;
  }

  // Epilogue. C layout: VGPR r -> row r + 8*lhalf, lane%16 -> col.
#pragma unroll
  for (int mt = 0; mt < 4; ++mt) {
#pragma unroll
    for (int nt = 0; nt < 2; ++nt) {
      const int col = tileN + waveN * 32 + nt * 16 + lmod;
      float bv = 0.f;
      if constexpr (EPI == EPI_BIAS_GELU_BF16 || EPI == EPI_BIAS_RESID_F32)
        bv = bias[col];
#pragma unroll
      for (int r = 0; r < 8; ++r) {
        const int row = tileM + waveM * 64 + mt * 16 + lhalf * 8 + r;
        const size_t idx = (size_t)row * N + col;
        float v = acc[mt][nt][r] + bv;
        if constexpr (EPI == EPI_BIAS_GELU_BF16)
          v = 0.5f * v * (1.0f + erff(v * 0.70710678118654752f));
        if constexpr (EPI == EPI_RESID_F32 || EPI == EPI_BIAS_RESID_F32)
          v += resid[idx];
        if constexpr (EPI == EPI_STORE_BF16 || EPI == EPI_BIAS_GELU_BF16)
          outB[idx] = f2bf(v);
        else
          outF[idx] = v;
      }
    }
  }
}

// ------------------------------- attention ---------------------------------
// One block = (b, h, 128 q rows); 256 threads = 8 waves, 16 q rows per wave.
// Flash loop over 64-wide key chunks with fused alibi and online softmax.
__global__ __launch_bounds__(256) void attn_kernel(
    const unsigned short* __restrict__ qkvb,  // [B*T, 3*H*Dh] bf16
    const float* __restrict__ alibi,          // [H, T, T] fp32
    unsigned short* __restrict__ yb,          // [B*T, H*Dh] bf16
    int T, int H) {
  const int Dh = 64;
  const int HD = H * Dh;
  const int E  = 3 * HD;

  __shared__ __align__(16) unsigned short kT[64 * LDSS];        // [t_k][d]
  __shared__ __align__(16) unsigned short vT[64 * LDSS];        // [d][t_k]
  __shared__ __align__(16) unsigned short pT[8][16 * LDSS];     // per-wave P

  const int nq = T / 128;
  const int blk = blockIdx.x;
  const int q0 = (blk % nq) * 128;
  const int bh = blk / nq;
  const int h = bh % H;
  const int b = bh / H;

  const int tid   = threadIdx.x;
  const int wave  = tid >> 5;
  const int lane  = tid & 31;
  const int lmod  = lane & 15;
  const int lhalf = lane >> 4;

  // Q fragments resident in registers (A layout: lane = q row).
  Frag qf[2];
  {
    int qrow = q0 + wave * 16 + lmod;
    const unsigned short* qp = qkvb + (size_t)(b * T + qrow) * E + h * Dh;
#pragma unroll
    for (int ks = 0; ks < 2; ++ks) {
      qf[ks].f[0] = *reinterpret_cast<const f4raw*>(qp + ks * 32 + lhalf * 8);
      qf[ks].f[1] = *reinterpret_cast<const f4raw*>(qp + ks * 32 + 16 + lhalf * 8);
    }
  }

  const v8f vzero = {0.f, 0.f, 0.f, 0.f, 0.f, 0.f, 0.f, 0.f};
  float m[8], l[8];
  v8f o[4];
#pragma unroll
  for (int r = 0; r < 8; ++r) { m[r] = -1e30f; l[r] = 0.f; }
#pragma unroll
  for (int nt = 0; nt < 4; ++nt) o[nt] = vzero;

  const float scale = 0.125f;  // 1/sqrt(64)
  const int rbase = q0 + wave * 16 + lhalf * 8;  // C-layout row base
  const float* arow0 = alibi + ((size_t)h * T + rbase) * T;

  for (int kc = 0; kc < T; kc += 64) {
    __syncthreads();
    // K chunk [64 x Dh], row-major (direct B-operand layout): async to LDS.
#pragma unroll
    for (int c = 0; c < 2; ++c) {
      int idx = tid + c * 256;
      int r = idx >> 3;
      int kk = (idx & 7) << 3;
      async_copy16(&kT[r * LDSS + kk],
                   &qkvb[(size_t)(b * T + kc + r) * E + HD + h * Dh + kk]);
    }
    // V chunk transposed to [d][t_k] so P@V uses the same B-operand layout.
#pragma unroll
    for (int c = 0; c < 16; ++c) {
      int idx = tid + c * 256;
      int t = idx >> 6;
      int d = idx & 63;
      vT[d * LDSS + t] = qkvb[(size_t)(b * T + kc + t) * E + 2 * HD + h * Dh + d];
    }
    // Prefetch next chunk's alibi rows into GL2 while we compute this chunk.
    if (lmod == 0 && kc + 64 < T) {
#pragma unroll
      for (int r = 0; r < 8; ++r)
        __builtin_prefetch(arow0 + (size_t)r * T + kc + 64, 0, 1);
    }
    async_wait();
    __syncthreads();

    // S = Q * K^T  (Dh=64 -> two 16x16x32 steps per N tile)
    v8f s[4];
#pragma unroll
    for (int nt = 0; nt < 4; ++nt) s[nt] = vzero;
#pragma unroll
    for (int ks = 0; ks < 2; ++ks)
#pragma unroll
      for (int nt = 0; nt < 4; ++nt) {
        Frag bfr;
        const unsigned short* p =
            &kT[(nt * 16 + lmod) * LDSS + ks * 32 + lhalf * 8];
        bfr.f[0] = *reinterpret_cast<const f4raw*>(p);
        bfr.f[1] = *reinterpret_cast<const f4raw*>(p + 16);
        s[nt] = __builtin_amdgcn_wmma_f32_16x16x32_bf16(
            false, qf[ks].v, false, bfr.v, (short)0, s[nt], false, false);
      }

    // Fuse alibi + scale; online softmax (row stats across 16-lane halves).
    float mc[8];
#pragma unroll
    for (int r = 0; r < 8; ++r) mc[r] = -1e30f;
    const float* arow = arow0 + kc;
#pragma unroll
    for (int nt = 0; nt < 4; ++nt) {
      int gcol = nt * 16 + lmod;
#pragma unroll
      for (int r = 0; r < 8; ++r) {
        float v = s[nt][r] * scale + arow[(size_t)r * T + gcol];
        s[nt][r] = v;
        mc[r] = fmaxf(mc[r], v);
      }
    }
    float rs[8];
#pragma unroll
    for (int r = 0; r < 8; ++r) {
      float mn = fmaxf(m[r], redmax16(mc[r]));
      float corr = __expf(m[r] - mn);
      m[r] = mn;
      l[r] *= corr;
#pragma unroll
      for (int nt = 0; nt < 4; ++nt) o[nt][r] *= corr;
      rs[r] = 0.f;
    }
#pragma unroll
    for (int nt = 0; nt < 4; ++nt)
#pragma unroll
      for (int r = 0; r < 8; ++r) {
        float p = __expf(s[nt][r] - m[r]);
        rs[r] += p;
        pT[wave][(r + lhalf * 8) * LDSS + nt * 16 + lmod] = f2bf(p);
      }
#pragma unroll
    for (int r = 0; r < 8; ++r) l[r] += redsum16(rs[r]);

    // P was written C-layout; re-read as A fragments (own wave region only).
    asm volatile("s_wait_dscnt 0" ::: "memory");
#pragma unroll
    for (int ks = 0; ks < 2; ++ks) {
      Frag af;
      const unsigned short* pa = &pT[wave][lmod * LDSS + ks * 32 + lhalf * 8];
      af.f[0] = *reinterpret_cast<const f4raw*>(pa);
      af.f[1] = *reinterpret_cast<const f4raw*>(pa + 16);
#pragma unroll
      for (int nt = 0; nt < 4; ++nt) {
        Frag bfr;
        const unsigned short* pb =
            &vT[(nt * 16 + lmod) * LDSS + ks * 32 + lhalf * 8];
        bfr.f[0] = *reinterpret_cast<const f4raw*>(pb);
        bfr.f[1] = *reinterpret_cast<const f4raw*>(pb + 16);
        o[nt] = __builtin_amdgcn_wmma_f32_16x16x32_bf16(
            false, af.v, false, bfr.v, (short)0, o[nt], false, false);
      }
    }
  }

#pragma unroll
  for (int nt = 0; nt < 4; ++nt)
#pragma unroll
    for (int r = 0; r < 8; ++r) {
      int grow = rbase + r;
      int gcol = h * Dh + nt * 16 + lmod;
      float v = o[nt][r] / l[r];
      yb[(size_t)(b * T + grow) * HD + gcol] = f2bf(v);
    }
}

// ------------------------------- layer norm --------------------------------
__global__ __launch_bounds__(256) void ln_bf16_kernel(
    const float* __restrict__ y, const float* __restrict__ w,
    unsigned short* __restrict__ out, int D) {
  int row = blockIdx.x;
  const float* yr = y + (size_t)row * D;
  __shared__ float red[8];
  __shared__ float red2[8];
  int tid = threadIdx.x;
  int lane = tid & 31, wv = tid >> 5;
  float vals[4];
  float sum = 0.f;
#pragma unroll
  for (int i = 0; i < 4; ++i) { vals[i] = yr[tid + i * 256]; sum += vals[i]; }
  sum = redsum16(sum);
  sum += __shfl_xor(sum, 16, 32);
  if (lane == 0) red[wv] = sum;
  __syncthreads();
  float tot = 0.f;
#pragma unroll
  for (int i = 0; i < 8; ++i) tot += red[i];
  float mu = tot * (1.0f / (float)D);
  float vsum = 0.f;
#pragma unroll
  for (int i = 0; i < 4; ++i) { float d = vals[i] - mu; vsum += d * d; }
  vsum = redsum16(vsum);
  vsum += __shfl_xor(vsum, 16, 32);
  if (lane == 0) red2[wv] = vsum;
  __syncthreads();
  float tv = 0.f;
#pragma unroll
  for (int i = 0; i < 8; ++i) tv += red2[i];
  float inv = rsqrtf(tv * (1.0f / (float)D) + 1e-5f);
#pragma unroll
  for (int i = 0; i < 4; ++i) {
    int col = tid + i * 256;
    out[(size_t)row * D + col] = f2bf((vals[i] - mu) * inv * w[col]);
  }
}

// ------------------------------- launcher ----------------------------------
extern "C" void kernel_launch(void* const* d_in, const int* in_sizes, int n_in,
                              void* d_out, int out_size, void* d_ws,
                              size_t ws_size, hipStream_t stream) {
  (void)in_sizes; (void)n_in; (void)out_size; (void)ws_size;

  const float* x         = (const float*)d_in[0];
  const float* alibi     = (const float*)d_in[1];
  // d_in[2] = ln1_w: unused (reference faithfully feeds raw x into QKV).
  const float* w_qkv     = (const float*)d_in[3];
  const float* w_out     = (const float*)d_in[4];
  const float* ln2_w     = (const float*)d_in[5];
  const float* w_mlp_in  = (const float*)d_in[6];
  const float* b_mlp_in  = (const float*)d_in[7];
  const float* w_mlp_out = (const float*)d_in[8];
  const float* b_mlp_out = (const float*)d_in[9];
  float* out = (float*)d_out;

  const int Bb = 2, T = 2048, D = 1024, H = 16, Dh = 64, Ff = 4096;
  const int BT = Bb * T;           // 4096
  const int HD = H * Dh;           // 1024
  const int E  = 3 * HD;           // 3072

  char* wp = (char*)d_ws;
  auto alloc = [&](size_t bytes) {
    char* p = wp;
    wp += (bytes + 255) & ~(size_t)255;
    return p;
  };
  unsigned short* xb       = (unsigned short*)alloc((size_t)BT * D * 2);
  unsigned short* wqkvb    = (unsigned short*)alloc((size_t)E * D * 2);
  unsigned short* woutb    = (unsigned short*)alloc((size_t)D * HD * 2);
  unsigned short* wmlpinb  = (unsigned short*)alloc((size_t)Ff * D * 2);
  unsigned short* wmlpoutb = (unsigned short*)alloc((size_t)D * Ff * 2);
  unsigned short* qkvb     = (unsigned short*)alloc((size_t)BT * E * 2);
  unsigned short* yb       = (unsigned short*)alloc((size_t)BT * HD * 2);
  float*          yf       = (float*)alloc((size_t)BT * D * 4);
  unsigned short* hb       = (unsigned short*)alloc((size_t)BT * D * 2);
  unsigned short* hf       = (unsigned short*)alloc((size_t)BT * Ff * 2);

  // 1) fp32 -> bf16 converts
  f32_to_bf16_kernel<<<512, 256, 0, stream>>>(x, xb, BT * D);
  f32_to_bf16_kernel<<<512, 256, 0, stream>>>(w_qkv, wqkvb, E * D);
  f32_to_bf16_kernel<<<512, 256, 0, stream>>>(w_out, woutb, D * HD);
  f32_to_bf16_kernel<<<512, 256, 0, stream>>>(w_mlp_in, wmlpinb, Ff * D);
  f32_to_bf16_kernel<<<512, 256, 0, stream>>>(w_mlp_out, wmlpoutb, D * Ff);

  // 2) QKV projection: [BT,E] = x[BT,D] * w_qkv[E,D]^T
  gemm_bf16_kernel<EPI_STORE_BF16><<<dim3(E / BN, BT / BM), 256, 0, stream>>>(
      xb, wqkvb, BT, E, D, nullptr, nullptr, nullptr, qkvb);

  // 3) Attention (fused alibi + softmax), yb = attn output [BT, HD]
  attn_kernel<<<Bb * H * (T / 128), 256, 0, stream>>>(qkvb, alibi, yb, T, H);

  // 4) Out projection + residual: yf = x + yb * w_out^T
  gemm_bf16_kernel<EPI_RESID_F32><<<dim3(D / BN, BT / BM), 256, 0, stream>>>(
      yb, woutb, BT, D, HD, nullptr, x, yf, nullptr);

  // 5) LN2 -> bf16
  ln_bf16_kernel<<<BT, 256, 0, stream>>>(yf, ln2_w, hb, D);

  // 6) MLP in + bias + exact GELU: hf = gelu(hb * w_mlp_in^T + b)
  gemm_bf16_kernel<EPI_BIAS_GELU_BF16>
      <<<dim3(Ff / BN, BT / BM), 256, 0, stream>>>(
          hb, wmlpinb, BT, Ff, D, b_mlp_in, nullptr, nullptr, hf);

  // 7) MLP out + bias + residual: out = yf + hf * w_mlp_out^T + b
  gemm_bf16_kernel<EPI_BIAS_RESID_F32>
      <<<dim3(D / BN, BT / BM), 256, 0, stream>>>(
          hf, wmlpoutb, BT, D, Ff, b_mlp_out, yf, out, nullptr);
}
